// Mamba2VisionMixer_29334626632152
// MI455X (gfx1250) — matrix operand order repaired and checked by reading, not compiled, not run on hardware
//
#include <hip/hip_runtime.h>

// ---------------- problem constants ----------------
#define BQ      2
#define LQ      4096
#define BL      8192        // BQ*LQ
#define DMODEL  1024
#define DINNER  2048
#define NH      32
#define HD      64
#define DSTATE  128
#define CONVD   2304        // DINNER + 2*DSTATE
#define NCOLS1  2336        // CONVD + NH  (only cols 2048.. of in_proj are used)
#define DPROJ   4384        // full in_proj cols
#define CHK     64          // chunk length
#define NCHK    64          // chunks per sequence

typedef __attribute__((ext_vector_type(16))) __bf16 bf16x16;
typedef __attribute__((ext_vector_type(8)))  float  floatx8;

// ---------------- small helpers ----------------
__device__ __forceinline__ __bf16 f2bf(float f) {
  union { float f; unsigned u; } v; v.f = f;
  unsigned r = v.u + 0x7FFFu + ((v.u >> 16) & 1u);   // round-to-nearest-even
  unsigned short h = (unsigned short)(r >> 16);
  __bf16 b; __builtin_memcpy(&b, &h, 2); return b;
}
__device__ __forceinline__ float bf2f(__bf16 b) {
  unsigned short h; __builtin_memcpy(&h, &b, 2);
  union { unsigned u; float f; } v; v.u = ((unsigned)h) << 16; return v.f;
}
__device__ __forceinline__ floatx8 vzero8() {
  floatx8 z;
#pragma unroll
  for (int r = 0; r < 8; ++r) z[r] = 0.f;
  return z;
}

// CDNA5 async copy: memory -> LDS without touching VGPRs (ASYNCcnt tracked).
__device__ __forceinline__ void async_ld16(void* lds, const void* g) {
  unsigned l = (unsigned)(size_t)lds;   // generic->LDS: low 32 bits are LDS offset
  asm volatile("global_load_async_to_lds_b128 %0, %1, off"
               :: "v"(l), "v"(g) : "memory");
}
__device__ __forceinline__ void wait_async0() {
  asm volatile("s_wait_asynccnt 0x0" ::: "memory");
}

// A fragment (16x32 bf16, MxK): lanes 0-15 -> M=lane, K = {0..7,16..23};
// lanes 16-31 -> M=lane-16, K = {8..15,24..31}.  (ISA 05_wmma 16-bit A layout)
__device__ __forceinline__ bf16x16 load_a_frag(const __bf16* base, int ld,
                                               int m0, int k0, int lane) {
  const __bf16* p = base + (size_t)(m0 + (lane & 15)) * ld + k0 + ((lane >> 4) << 3);
  bf16x16 a;
#pragma unroll
  for (int e = 0; e < 8; ++e) { a[e] = p[e]; a[e + 8] = p[e + 16]; }
  return a;
}
// B fragment (32x16 bf16, KxN) from an [N][K]-layout buffer:
// lane -> N = lane&15;  lanes 0-15 hold K={0..7,16..23}, lanes 16-31 hold K={8..15,24..31}.
__device__ __forceinline__ bf16x16 load_b_frag(const __bf16* base, int ld,
                                               int n0, int k0, int lane) {
  const __bf16* p = base + (size_t)(n0 + (lane & 15)) * ld + k0 + ((lane >> 4) << 3);
  bf16x16 b;
#pragma unroll
  for (int e = 0; e < 8; ++e) { b[e] = p[e]; b[e + 8] = p[e + 16]; }
  return b;
}
__device__ __forceinline__ floatx8 wmma_bf16(bf16x16 a, bf16x16 b, floatx8 c) {
  return __builtin_amdgcn_wmma_f32_16x16x32_bf16(false, a, false, b,
                                                 (short)0, c, false, false);
}

// ---------------- conversion kernels ----------------
__global__ void k_cvt_f32_bf16(const float* __restrict__ src,
                               __bf16* __restrict__ dst, long n) {
  long i = (long)blockIdx.x * blockDim.x + threadIdx.x;
  if (i < n) dst[i] = f2bf(src[i]);
}
// in_proj_w[:, 2048:4384] -> transposed bf16 [N=2336][K=1024]
__global__ void k_cvt_w1t(const float* __restrict__ w, __bf16* __restrict__ dst) {
  long i = (long)blockIdx.x * blockDim.x + threadIdx.x;
  const long n = (long)NCOLS1 * DMODEL;
  if (i >= n) return;
  long c = i / DMODEL, k = i - c * DMODEL;       // dst[c][k]
  dst[i] = f2bf(w[k * DPROJ + DINNER + c]);
}
// out_proj_w (2048x1024) -> transposed bf16 [N=1024][K=2048]
__global__ void k_cvt_w2t(const float* __restrict__ w, __bf16* __restrict__ dst) {
  long i = (long)blockIdx.x * blockDim.x + threadIdx.x;
  const long n = (long)DMODEL * DINNER;
  if (i >= n) return;
  long c = i / DINNER, k = i - c * DINNER;       // dst[c][k]
  dst[i] = f2bf(w[k * DMODEL + c]);
}

// ---- WMMA GEMM: C(f32,MxN) = A(bf16,MxK) @ Bt(bf16,[N][K])^T ----
// 256 threads = 8 waves; workgroup tile 128x128; wave tile 32x64; K-step 32.
// Double-buffered LDS staged via GLOBAL_LOAD_ASYNC_TO_LDS_B128.
__global__ void __launch_bounds__(256)
k_gemm_bf16(const __bf16* __restrict__ A, const __bf16* __restrict__ Bt,
            float* __restrict__ C, int M, int N, int K) {
  __shared__ __bf16 sA[2][128 * 40];   // [m][k], 80B padded rows (16B aligned)
  __shared__ __bf16 sB[2][128 * 40];   // [n][k]
  const int m0 = blockIdx.x * 128;
  const int n0 = blockIdx.y * 128;
  const int tid = threadIdx.x, lane = tid & 31, w = tid >> 5;
  const int wr = w >> 1, wc = w & 1;            // wave grid 4x2
  const int sr = tid >> 1, sc = (tid & 1) * 16; // staging: row, col half

  floatx8 acc[2][4];
#pragma unroll
  for (int i = 0; i < 2; ++i)
#pragma unroll
    for (int j = 0; j < 4; ++j) acc[i][j] = vzero8();

  const __bf16* gA = A + (size_t)(m0 + sr) * K + sc;
  const __bf16* gB = (n0 + sr < N) ? (Bt + (size_t)(n0 + sr) * K + sc)
                                   : (Bt + sc);   // clamp OOB rows (unused)
  const int nsteps = K / 32;

  // prologue: stage k-step 0 into buffer 0
  async_ld16(&sA[0][sr * 40 + sc], gA);
  async_ld16(&sB[0][sr * 40 + sc], gB);
  wait_async0();
  __syncthreads();

  for (int s = 0; s < nsteps; ++s) {
    const int cur = s & 1;
    if (s + 1 < nsteps) {            // prefetch next k-step into other buffer
      const int nxt = cur ^ 1;
      async_ld16(&sA[nxt][sr * 40 + sc], gA + (size_t)(s + 1) * 32);
      async_ld16(&sB[nxt][sr * 40 + sc], gB + (size_t)(s + 1) * 32);
    }
    bf16x16 af[2], bfv[4];
#pragma unroll
    for (int i = 0; i < 2; ++i)
      af[i] = load_a_frag(sA[cur], 40, wr * 32 + i * 16, 0, lane);
#pragma unroll
    for (int j = 0; j < 4; ++j)
      bfv[j] = load_b_frag(sB[cur], 40, wc * 64 + j * 16, 0, lane);
#pragma unroll
    for (int i = 0; i < 2; ++i)
#pragma unroll
      for (int j = 0; j < 4; ++j) acc[i][j] = wmma_bf16(af[i], bfv[j], acc[i][j]);
    wait_async0();                   // next buffer landed
    __syncthreads();                 // everyone done reading cur buffer
  }
  const int hi = lane >> 4, col = lane & 15;
#pragma unroll
  for (int i = 0; i < 2; ++i)
#pragma unroll
    for (int j = 0; j < 4; ++j) {
      int gm = m0 + wr * 32 + i * 16 + hi * 8;
      int gn = n0 + wc * 64 + j * 16 + col;
      if (gn < N) {
#pragma unroll
        for (int r = 0; r < 8; ++r)
          C[(size_t)(gm + r) * N + gn] = acc[i][j][r];
      }
    }
}

// ---------------- conv(4, depthwise, causal) + SiLU + softplus(dt) ----------------
__global__ void k_conv(const float* __restrict__ zx, const float* __restrict__ conv_w,
                       const float* __restrict__ conv_b, const float* __restrict__ dt_bias,
                       const float* __restrict__ A_log,
                       __bf16* __restrict__ xdtb, float* __restrict__ xhbuf,
                       __bf16* __restrict__ Bbf, __bf16* __restrict__ Cbf,
                       float* __restrict__ dAbuf) {
  long gid = (long)blockIdx.x * 256 + threadIdx.x;
  const long total = (long)BL * CONVD;
  if (gid >= total) return;
  int  ch = (int)(gid % CONVD);
  long bl = gid / CONVD;
  int  l = (int)(bl % LQ);
  int  b = (int)(bl / LQ);
  float acc = conv_b[ch];
#pragma unroll
  for (int j = 0; j < 4; ++j) {
    int ls = l - 3 + j;
    if (ls >= 0) acc += zx[((long)b * LQ + ls) * NCOLS1 + ch] * conv_w[j * CONVD + ch];
  }
  float sil = acc / (1.f + __expf(-acc));
  if (ch < DINNER) {
    int h = ch >> 6, pp = ch & 63;
    float dtraw = zx[bl * NCOLS1 + CONVD + h] + dt_bias[h];
    float dt = (dtraw > 20.f) ? dtraw : log1pf(__expf(dtraw));
    long seq = (long)b * NH + h;
    long o = (seq * LQ + l) * HD + pp;
    xhbuf[o] = sil;
    xdtb[o]  = f2bf(sil * dt);
    if (pp == 0) dAbuf[seq * LQ + l] = -__expf(A_log[h]) * dt;
  } else if (ch < DINNER + DSTATE) {
    Bbf[bl * DSTATE + (ch - DINNER)] = f2bf(sil);
  } else {
    Cbf[bl * DSTATE + (ch - DINNER - DSTATE)] = f2bf(sil);
  }
}

// ---------------- SSD chunked scan (1 workgroup per (b,h)) ----------------
// State S[p=64][n=128] kept in registers as WMMA accumulators (4 tiles/wave).
__global__ void __launch_bounds__(256)
k_ssd(const __bf16* __restrict__ Bbf, const __bf16* __restrict__ Cbf,
      const __bf16* __restrict__ xdtb, const float* __restrict__ xhbuf,
      const float* __restrict__ dAbuf, const float* __restrict__ Dvec,
      __bf16* __restrict__ ybf) {
  __shared__ __bf16 sBt[DSTATE * CHK];   // [n][l]   (B transposed)
  __shared__ __bf16 sXdtT[HD * CHK];     // [p][l]   (x*dt transposed)
  __shared__ __bf16 sXdecT[HD * CHK];    // [p][l]   (x*dt*decay_states, transposed)
  __shared__ __bf16 sM1[CHK * CHK];      // [l][s]   G .* L
  __shared__ __bf16 sSb[HD * DSTATE];    // [p][n]   bf16 snapshot of state
  __shared__ float  scum[CHK];           // inclusive cumsum of dA

  const int bh = blockIdx.x;             // 0..63
  const int b = bh >> 5, h = bh & 31;
  const int tid = threadIdx.x, lane = tid & 31, w = tid >> 5;
  const int hi = lane >> 4, cc = lane & 15;
  const int tm = w >> 1;                 // wave M-tile (0..3)
  const int wn = w & 1;
  const float Dh = Dvec[h];

  const __bf16* Bbase = Bbf  + (size_t)b * LQ * DSTATE;
  const __bf16* Cbase = Cbf  + (size_t)b * LQ * DSTATE;
  const __bf16* Xbase = xdtb + (size_t)bh * LQ * HD;
  const float*  XHb   = xhbuf + (size_t)bh * LQ * HD;
  const float*  dAb   = dAbuf + (size_t)bh * LQ;

  floatx8 Sreg[4];
#pragma unroll
  for (int t = 0; t < 4; ++t) Sreg[t] = vzero8();

  for (int ci = 0; ci < NCHK; ++ci) {
    const int l0 = ci * CHK;
    const __bf16* Bc = Bbase + (size_t)l0 * DSTATE;
    const __bf16* Cc = Cbase + (size_t)l0 * DSTATE;
    const __bf16* Xc = Xbase + (size_t)l0 * HD;

    // prefetch next chunk's operands into cache (global_prefetch_b8)
    if (ci + 1 < NCHK) {
      __builtin_prefetch(Bc + CHK * DSTATE + (size_t)tid * 32, 0, 0);
      __builtin_prefetch(Cc + CHK * DSTATE + (size_t)tid * 32, 0, 0);
      __builtin_prefetch(Xc + CHK * HD + (size_t)tid * 16, 0, 0);
    }

    // ---- phase A: cumsum(dA) + stage Bt and XdtT into LDS ----
    if (tid == 0) {
      float run = 0.f;
      for (int l = 0; l < CHK; ++l) { run += dAb[l0 + l]; scum[l] = run; }
    }
    { int l = tid & 63, nb = (tid >> 6) * 32;
      const __bf16* g = Bc + (size_t)l * DSTATE + nb;
      for (int e = 0; e < 32; ++e) sBt[(nb + e) * CHK + l] = g[e];
    }
    { int l = tid & 63, pb = (tid >> 6) * 16;
      const __bf16* g = Xc + (size_t)l * HD + pb;
      for (int e = 0; e < 16; ++e) sXdtT[(pb + e) * CHK + l] = g[e];
    }
    __syncthreads();

    const float Atot = scum[CHK - 1];

    // ---- phase B: Xdec, state snapshot, G = C*B^T masked by L ----
    { int l = tid & 63, pb = (tid >> 6) * 16;
      float dec = __expf(Atot - scum[l]);
      for (int e = 0; e < 16; ++e)
        sXdecT[(pb + e) * CHK + l] = f2bf(bf2f(sXdtT[(pb + e) * CHK + l]) * dec);
    }
#pragma unroll
    for (int t = 0; t < 4; ++t) {      // snapshot S -> bf16 LDS (D-frag layout)
      int nB = wn * 64 + t * 16 + cc;
      int pB = tm * 16 + hi * 8;
#pragma unroll
      for (int r = 0; r < 8; ++r) sSb[(size_t)(pB + r) * DSTATE + nB] = f2bf(Sreg[t][r]);
    }
#pragma unroll
    for (int t = 0; t < 2; ++t) {      // G tiles (A,B fragments straight from global)
      int tn = wn * 2 + t;
      floatx8 g = vzero8();
      for (int kk = 0; kk < DSTATE; kk += 32) {
        bf16x16 af = load_a_frag(Cc, DSTATE, tm * 16, kk, lane);
        bf16x16 bf = load_b_frag(Bc, DSTATE, tn * 16, kk, lane);
        g = wmma_bf16(af, bf, g);
      }
      int lrow0 = tm * 16 + hi * 8, scol = tn * 16 + cc;
#pragma unroll
      for (int r = 0; r < 8; ++r) {
        int lrow = lrow0 + r;
        float v = (lrow >= scol) ? g[r] * __expf(scum[lrow] - scum[scol]) : 0.f;
        sM1[lrow * CHK + scol] = f2bf(v);
      }
    }
    __syncthreads();

    // ---- phase C: Y = M1*xdt + exp(cumA)*(C*S^T) + D*xh ; state update ----
#pragma unroll
    for (int t = 0; t < 2; ++t) {
      int tn = wn * 2 + t;
      floatx8 yd = vzero8(), yo = vzero8();
      for (int kk = 0; kk < CHK; kk += 32) {
        bf16x16 af = load_a_frag(sM1, CHK, tm * 16, kk, lane);
        bf16x16 bf = load_b_frag(sXdtT, CHK, tn * 16, kk, lane);
        yd = wmma_bf16(af, bf, yd);
      }
      for (int kk = 0; kk < DSTATE; kk += 32) {
        bf16x16 af = load_a_frag(Cc, DSTATE, tm * 16, kk, lane);
        bf16x16 bf = load_b_frag(sSb, DSTATE, tn * 16, kk, lane);
        yo = wmma_bf16(af, bf, yo);
      }
      int lrow0 = tm * 16 + hi * 8, pc = tn * 16 + cc;
#pragma unroll
      for (int r = 0; r < 8; ++r) {
        int lrow = lrow0 + r;
        float y = yd[r] + __expf(scum[lrow]) * yo[r]
                + Dh * XHb[(size_t)(l0 + lrow) * HD + pc];
        ybf[((size_t)(b * LQ + l0 + lrow)) * DINNER + h * HD + pc] = f2bf(y);
      }
    }
    { // S <- exp(Atot)*S + Xdec^T @ B
      float cdec = __expf(Atot);
#pragma unroll
      for (int t = 0; t < 4; ++t)
#pragma unroll
        for (int r = 0; r < 8; ++r) Sreg[t][r] *= cdec;
      for (int kk = 0; kk < CHK; kk += 32) {
        bf16x16 af = load_a_frag(sXdecT, CHK, tm * 16, kk, lane);
#pragma unroll
        for (int t = 0; t < 4; ++t) {
          bf16x16 bf = load_b_frag(sBt, CHK, (wn * 4 + t) * 16, kk, lane);
          Sreg[t] = wmma_bf16(af, bf, Sreg[t]);
        }
      }
    }
    __syncthreads();
  }
}

// ---------------- host launcher ----------------
extern "C" void kernel_launch(void* const* d_in, const int* in_sizes, int n_in,
                              void* d_out, int out_size, void* d_ws, size_t ws_size,
                              hipStream_t stream) {
  const float* x         = (const float*)d_in[0];
  const float* in_proj_w = (const float*)d_in[1];
  const float* conv_w    = (const float*)d_in[2];
  const float* conv_b    = (const float*)d_in[3];
  const float* dt_bias   = (const float*)d_in[4];
  const float* A_log     = (const float*)d_in[5];
  const float* Dvec      = (const float*)d_in[6];
  const float* out_w     = (const float*)d_in[7];
  float* out = (float*)d_out;

  // workspace carve-out (256B aligned slices)
  size_t off = 0;
  auto take = [&](size_t bytes) -> char* {
    char* r = (char*)d_ws + off;
    off = (off + bytes + 255) & ~(size_t)255;
    return r;
  };
  __bf16* xb16  = (__bf16*)take((size_t)BL * DMODEL * 2);
  __bf16* w1t   = (__bf16*)take((size_t)NCOLS1 * DMODEL * 2);  // [2336][1024]
  __bf16* w2t   = (__bf16*)take((size_t)DMODEL * DINNER * 2);  // [1024][2048]
  float*  zx    = (float*) take((size_t)BL * NCOLS1 * 4);
  __bf16* xdtb  = (__bf16*)take((size_t)BQ * NH * LQ * HD * 2);
  float*  xhbuf = (float*) take((size_t)BQ * NH * LQ * HD * 4);
  __bf16* Bbf   = (__bf16*)take((size_t)BL * DSTATE * 2);
  __bf16* Cbf   = (__bf16*)take((size_t)BL * DSTATE * 2);
  float*  dAbuf = (float*) take((size_t)BQ * NH * LQ * 4);
  __bf16* ybf   = (__bf16*)take((size_t)BL * DINNER * 2);
  (void)ws_size; (void)in_sizes; (void)n_in; (void)out_size;

  // 1) precision conversions (+ weight transposes to [N][K])
  { long n = (long)BL * DMODEL;
    k_cvt_f32_bf16<<<(unsigned)((n + 255) / 256), 256, 0, stream>>>(x, xb16, n); }
  { long n = (long)NCOLS1 * DMODEL;
    k_cvt_w1t<<<(unsigned)((n + 255) / 256), 256, 0, stream>>>(in_proj_w, w1t); }
  { long n = (long)DMODEL * DINNER;
    k_cvt_w2t<<<(unsigned)((n + 255) / 256), 256, 0, stream>>>(out_w, w2t); }

  // 2) in_proj GEMM (only the xbc+dt columns): 8192x2336 = x(8192x1024) @ W1
  k_gemm_bf16<<<dim3(BL / 128, (NCOLS1 + 127) / 128), 256, 0, stream>>>(
      xb16, w1t, zx, BL, NCOLS1, DMODEL);

  // 3) depthwise causal conv + SiLU + softplus(dt); emit SSD operands
  { long total = (long)BL * CONVD;
    k_conv<<<(unsigned)((total + 255) / 256), 256, 0, stream>>>(
        zx, conv_w, conv_b, dt_bias, A_log, xdtb, xhbuf, Bbf, Cbf, dAbuf); }

  // 4) SSD chunked scan, one workgroup per (batch, head)
  k_ssd<<<BQ * NH, 256, 0, stream>>>(Bbf, Cbf, xdtb, xhbuf, dAbuf, Dvec, ybf);

  // 5) out_proj GEMM -> fp32 output
  k_gemm_bf16<<<dim3(BL / 128, DMODEL / 128), 256, 0, stream>>>(
      ybf, w2t, out, BL, DMODEL, DINNER);
}
